// RecurrentRelationalNetwork_8967891714099
// MI455X (gfx1250) — matrix operand order repaired and checked by reading, not compiled, not run on hardware
//
#include <hip/hip_runtime.h>

typedef __attribute__((ext_vector_type(16))) __bf16 v16bf;
typedef __attribute__((ext_vector_type(8)))  __bf16 v8bf;
typedef __attribute__((ext_vector_type(8)))  float  v8f;

#define WAVES_PER_BLOCK 4

// ---------------------------------------------------------------------------
// Edge kernel: gather h[src]||h[dst] -> 3-layer MLP on WMMA -> atomic scatter
// One wave32 handles a tile of 16 edges. bf16 operands, f32 accumulation.
// ---------------------------------------------------------------------------
__global__ __launch_bounds__(128)
void edge_mlp_scatter(const float* __restrict__ h,
                      const int*   __restrict__ src,
                      const int*   __restrict__ dst,
                      const float* __restrict__ W1, const float* __restrict__ b1,
                      const float* __restrict__ W2, const float* __restrict__ b2,
                      const float* __restrict__ W3, const float* __restrict__ b3,
                      float* __restrict__ agg,
                      int E)
{
    // Weights pre-packed into WMMA B-operand layout:
    //   B[k][n] lives at tile*512 + lane*16 + (k&7) + 8*(k_in_tile>=16), lane = (n&15) + 16*((k_in_tile>>3)&1)
    __shared__ __align__(32) __bf16 sW1[6 * 512];        //  32x96 -> 6 tiles
    __shared__ __align__(32) __bf16 sW2[18 * 512];       //  96x96 -> 3x6 tiles
    __shared__ __align__(32) __bf16 sW3[3 * 512];        //  96x16 -> 3 tiles
    __shared__ float sb1[96], sb2[96], sb3[16];
    __shared__ __align__(32) __bf16 sHa[WAVES_PER_BLOCK][16 * 96]; // layer-1 out
    __shared__ __align__(32) __bf16 sHb[WAVES_PER_BLOCK][16 * 96]; // layer-2 out
    __shared__ int sDst[WAVES_PER_BLOCK][16];

    const int tid = threadIdx.x;

    for (int idx = tid; idx < 32 * 96; idx += blockDim.x) {
        int k = idx / 96, n = idx % 96;
        int nt = n >> 4;
        int lane = (n & 15) + ((k & 8) ? 16 : 0);
        sW1[nt * 512 + lane * 16 + (k & 7) + ((k >= 16) ? 8 : 0)] = (__bf16)W1[idx];
    }
    for (int idx = tid; idx < 96 * 96; idx += blockDim.x) {
        int k = idx / 96, n = idx % 96;
        int kt = k / 32, kk = k & 31;
        int nt = n >> 4;
        int lane = (n & 15) + ((kk & 8) ? 16 : 0);
        sW2[(kt * 6 + nt) * 512 + lane * 16 + (kk & 7) + ((kk >= 16) ? 8 : 0)] = (__bf16)W2[idx];
    }
    for (int idx = tid; idx < 96 * 16; idx += blockDim.x) {
        int k = idx / 16, n = idx % 16;
        int kt = k / 32, kk = k & 31;
        int lane = n + ((kk & 8) ? 16 : 0);
        sW3[kt * 512 + lane * 16 + (kk & 7) + ((kk >= 16) ? 8 : 0)] = (__bf16)W3[idx];
    }
    if (tid < 96) { sb1[tid] = b1[tid]; sb2[tid] = b2[tid]; }
    if (tid < 16) { sb3[tid] = b3[tid]; }
    __syncthreads();

    const int wave = tid >> 5;
    const int lane = tid & 31;
    const int r    = lane & 15;               // edge row within tile / N column
    const int off  = (lane < 16) ? 0 : 8;     // A-layout K sub-offset per half-wave
    const int mb   = (lane < 16) ? 0 : 8;     // C-layout row base per half-wave
    const int ncol = r;

    const int ntiles  = (E + 15) >> 4;
    const int wstride = gridDim.x * WAVES_PER_BLOCK;

    for (int tile = blockIdx.x * WAVES_PER_BLOCK + wave; tile < ntiles; tile += wstride) {
        const int ebase = tile << 4;
        int e = ebase + r;
        if (e >= E) e = E - 1;
        const int si = src[e];
        const int di = dst[e];
        if (lane < 16) sDst[wave][r] = di;
        __asm volatile("" ::: "memory");

        // ---- assemble A0 (16 edges x 32 feats = h_src||h_dst) in WMMA A layout
        const float* hs = h + (size_t)si * 16 + off;
        const float* hd = h + (size_t)di * 16 + off;
        float4 s0 = *(const float4*)(hs);
        float4 s1 = *(const float4*)(hs + 4);
        float4 d0 = *(const float4*)(hd);
        float4 d1 = *(const float4*)(hd + 4);
        v16bf a0;
        a0[0]  = (__bf16)s0.x; a0[1]  = (__bf16)s0.y; a0[2]  = (__bf16)s0.z; a0[3]  = (__bf16)s0.w;
        a0[4]  = (__bf16)s1.x; a0[5]  = (__bf16)s1.y; a0[6]  = (__bf16)s1.z; a0[7]  = (__bf16)s1.w;
        a0[8]  = (__bf16)d0.x; a0[9]  = (__bf16)d0.y; a0[10] = (__bf16)d0.z; a0[11] = (__bf16)d0.w;
        a0[12] = (__bf16)d1.x; a0[13] = (__bf16)d1.y; a0[14] = (__bf16)d1.z; a0[15] = (__bf16)d1.w;

        // ---- layer 1: [16x32] @ [32x96] + b1, relu
        #pragma unroll
        for (int nt = 0; nt < 6; ++nt) {
            v16bf b = *(const v16bf*)(&sW1[nt * 512 + lane * 16]);
            v8f c = {};
            c = __builtin_amdgcn_wmma_f32_16x16x32_bf16(false, a0, false, b,
                                                        (short)0, c, false, false);
            float bias = sb1[nt * 16 + ncol];
            #pragma unroll
            for (int i = 0; i < 8; ++i) {
                float v = c[i] + bias;
                v = v > 0.f ? v : 0.f;
                sHa[wave][(mb + i) * 96 + nt * 16 + ncol] = (__bf16)v;
            }
        }
        __asm volatile("s_wait_dscnt 0" ::: "memory");

        // ---- layer 2: [16x96] @ [96x96] + b2, relu
        v16bf a2[3];
        #pragma unroll
        for (int kt = 0; kt < 3; ++kt) {
            const __bf16* p = &sHa[wave][r * 96 + kt * 32];
            v8bf lo = *(const v8bf*)(p + off);
            v8bf hi = *(const v8bf*)(p + 16 + off);
            a2[kt] = __builtin_shufflevector(lo, hi,
                       0,1,2,3,4,5,6,7,8,9,10,11,12,13,14,15);
        }
        __asm volatile("" ::: "memory");
        #pragma unroll
        for (int nt = 0; nt < 6; ++nt) {
            v8f c = {};
            #pragma unroll
            for (int kt = 0; kt < 3; ++kt) {
                v16bf b = *(const v16bf*)(&sW2[(kt * 6 + nt) * 512 + lane * 16]);
                c = __builtin_amdgcn_wmma_f32_16x16x32_bf16(false, a2[kt], false, b,
                                                            (short)0, c, false, false);
            }
            float bias = sb2[nt * 16 + ncol];
            #pragma unroll
            for (int i = 0; i < 8; ++i) {
                float v = c[i] + bias;
                v = v > 0.f ? v : 0.f;
                sHb[wave][(mb + i) * 96 + nt * 16 + ncol] = (__bf16)v;
            }
        }
        __asm volatile("s_wait_dscnt 0" ::: "memory");

        // ---- layer 3: [16x96] @ [96x16] + b3
        v8f c3 = {};
        #pragma unroll
        for (int kt = 0; kt < 3; ++kt) {
            const __bf16* p = &sHb[wave][r * 96 + kt * 32];
            v8bf lo = *(const v8bf*)(p + off);
            v8bf hi = *(const v8bf*)(p + 16 + off);
            v16bf a = __builtin_shufflevector(lo, hi,
                        0,1,2,3,4,5,6,7,8,9,10,11,12,13,14,15);
            v16bf b = *(const v16bf*)(&sW3[kt * 512 + lane * 16]);
            c3 = __builtin_amdgcn_wmma_f32_16x16x32_bf16(false, a, false, b,
                                                         (short)0, c3, false, false);
        }
        const float bias3 = sb3[ncol];

        // ---- scatter-sum into agg[dst] (L2-resident, fp32 atomics)
        #pragma unroll
        for (int i = 0; i < 8; ++i) {
            int erow = mb + i;
            if (ebase + erow < E) {
                int dn = sDst[wave][erow];
                __hip_atomic_fetch_add(&agg[(size_t)dn * 16 + ncol], c3[i] + bias3,
                                       __ATOMIC_RELAXED, __HIP_MEMORY_SCOPE_AGENT);
            }
        }
    }
}

// ---------------------------------------------------------------------------
// Node kernel: Keras GRUCell (reset_after=True). One thread per node.
// ---------------------------------------------------------------------------
__global__ __launch_bounds__(256)
void gru_update(const float* __restrict__ hin,
                const float* __restrict__ clues,
                const float* __restrict__ agg,
                const float* __restrict__ gk,    // [26,48]
                const float* __restrict__ grk,   // [16,48]
                const float* __restrict__ gb,    // [2,48]
                float* __restrict__ hout, int N)
{
    __shared__ float sK[26 * 48];
    __shared__ float sR[16 * 48];
    __shared__ float sB[96];
    for (int i = threadIdx.x; i < 26 * 48; i += blockDim.x) sK[i] = gk[i];
    for (int i = threadIdx.x; i < 16 * 48; i += blockDim.x) sR[i] = grk[i];
    for (int i = threadIdx.x; i < 96;      i += blockDim.x) sB[i] = gb[i];
    __syncthreads();

    int n = blockIdx.x * blockDim.x + threadIdx.x;
    if (n >= N) return;

    float x[26], hv[16];
    #pragma unroll
    for (int j = 0; j < 10; ++j) x[j] = clues[(size_t)n * 10 + j];
    #pragma unroll
    for (int j = 0; j < 16; ++j) x[10 + j] = agg[(size_t)n * 16 + j];
    #pragma unroll
    for (int j = 0; j < 16; ++j) hv[j] = hin[(size_t)n * 16 + j];

    float gz[16], gr[16];
    #pragma unroll
    for (int j = 0; j < 16; ++j) { gz[j] = sB[j] + sB[48 + j]; gr[j] = sB[16 + j] + sB[64 + j]; }
    #pragma unroll
    for (int k = 0; k < 26; ++k) {
        float xv = x[k]; const float* w = &sK[k * 48];
        #pragma unroll
        for (int j = 0; j < 16; ++j) { gz[j] += xv * w[j]; gr[j] += xv * w[16 + j]; }
    }
    #pragma unroll
    for (int k = 0; k < 16; ++k) {
        float hk = hv[k]; const float* w = &sR[k * 48];
        #pragma unroll
        for (int j = 0; j < 16; ++j) { gz[j] += hk * w[j]; gr[j] += hk * w[16 + j]; }
    }

    float xh[16], rh[16];
    #pragma unroll
    for (int j = 0; j < 16; ++j) { xh[j] = sB[32 + j]; rh[j] = sB[80 + j]; }
    #pragma unroll
    for (int k = 0; k < 26; ++k) {
        float xv = x[k]; const float* w = &sK[k * 48 + 32];
        #pragma unroll
        for (int j = 0; j < 16; ++j) xh[j] += xv * w[j];
    }
    #pragma unroll
    for (int k = 0; k < 16; ++k) {
        float hk = hv[k]; const float* w = &sR[k * 48 + 32];
        #pragma unroll
        for (int j = 0; j < 16; ++j) rh[j] += hk * w[j];
    }

    #pragma unroll
    for (int j = 0; j < 16; ++j) {
        float z  = 1.f / (1.f + __expf(-gz[j]));
        float rr = 1.f / (1.f + __expf(-gr[j]));
        float hh = tanhf(xh[j] + rr * rh[j]);
        hout[(size_t)n * 16 + j] = z * hv[j] + (1.f - z) * hh;
    }
}

// ---------------------------------------------------------------------------
// Output: softmax(h @ Wo + bo), one thread per node.
// ---------------------------------------------------------------------------
__global__ __launch_bounds__(256)
void out_softmax(const float* __restrict__ hin,
                 const float* __restrict__ Wo, const float* __restrict__ bo,
                 float* __restrict__ out, int N)
{
    int n = blockIdx.x * blockDim.x + threadIdx.x;
    if (n >= N) return;
    float hv[16];
    #pragma unroll
    for (int j = 0; j < 16; ++j) hv[j] = hin[(size_t)n * 16 + j];
    float lg[9];
    #pragma unroll
    for (int o = 0; o < 9; ++o) {
        float a = bo[o];
        #pragma unroll
        for (int j = 0; j < 16; ++j) a += hv[j] * Wo[j * 9 + o];
        lg[o] = a;
    }
    float m = lg[0];
    #pragma unroll
    for (int o = 1; o < 9; ++o) m = lg[o] > m ? lg[o] : m;
    float s = 0.f;
    #pragma unroll
    for (int o = 0; o < 9; ++o) { lg[o] = __expf(lg[o] - m); s += lg[o]; }
    float inv = 1.f / s;
    #pragma unroll
    for (int o = 0; o < 9; ++o) out[(size_t)n * 9 + o] = lg[o] * inv;
}

__global__ void fill_zero(float* __restrict__ p, long long n)
{
    long long i = (long long)blockIdx.x * blockDim.x + threadIdx.x;
    if (i < n) p[i] = 0.f;
}

__global__ void copy_f32(const float* __restrict__ a, float* __restrict__ b, long long n)
{
    long long i = (long long)blockIdx.x * blockDim.x + threadIdx.x;
    if (i < n) b[i] = a[i];
}

// ---------------------------------------------------------------------------
extern "C" void kernel_launch(void* const* d_in, const int* in_sizes, int n_in,
                              void* d_out, int out_size, void* d_ws, size_t ws_size,
                              hipStream_t stream)
{
    const float* clues  = (const float*)d_in[0];
    const float* h_init = (const float*)d_in[1];
    const int*   src    = (const int*)  d_in[2];
    const int*   dst    = (const int*)  d_in[3];
    const float* W1     = (const float*)d_in[4];
    const float* b1     = (const float*)d_in[5];
    const float* W2     = (const float*)d_in[6];
    const float* b2     = (const float*)d_in[7];
    const float* W3     = (const float*)d_in[8];
    const float* b3     = (const float*)d_in[9];
    const float* gk     = (const float*)d_in[10];
    const float* grk    = (const float*)d_in[11];
    const float* gb     = (const float*)d_in[12];
    const float* Wo     = (const float*)d_in[13];
    const float* bo     = (const float*)d_in[14];

    const int N = in_sizes[1] / 16;
    const int E = in_sizes[2];

    float* agg = (float*)d_ws;                     // [N,16]
    float* h0  = agg + (size_t)N * 16;             // [N,16]
    float* h1  = h0  + (size_t)N * 16;             // [N,16]
    const long long hn = (long long)N * 16;

    copy_f32<<<(int)((hn + 255) / 256), 256, 0, stream>>>(h_init, h0, hn);

    const int ntiles  = (E + 15) / 16;
    int eblocks = (ntiles + WAVES_PER_BLOCK - 1) / WAVES_PER_BLOCK;
    if (eblocks > 2048) eblocks = 2048;
    const int nblocks = (N + 255) / 256;

    float* hc  = h0;
    float* hnx = h1;
    for (int s = 0; s < 8; ++s) {
        fill_zero<<<(int)((hn + 255) / 256), 256, 0, stream>>>(agg, hn);
        edge_mlp_scatter<<<eblocks, 128, 0, stream>>>(hc, src, dst,
                                                      W1, b1, W2, b2, W3, b3,
                                                      agg, E);
        gru_update<<<nblocks, 256, 0, stream>>>(hc, clues, agg, gk, grk, gb, hnx, N);
        float* t = hc; hc = hnx; hnx = t;
    }
    out_softmax<<<nblocks, 256, 0, stream>>>(hc, Wo, bo, (float*)d_out, N);
}
